// SparseMultiHeadAttention_83073257439255
// MI455X (gfx1250) — compile-verified, matrix operand order checked by reading, not run on hardware
//
#include <hip/hip_runtime.h>
#include <hip/hip_bf16.h>

// Problem constants
#define NTOK 4096
#define CDIM 1024
#define HDS  16
#define DH   64
#define C3   3072

typedef __attribute__((ext_vector_type(16))) __bf16 v16bf;
typedef __attribute__((ext_vector_type(8)))  float  v8f;

union FragBF { v16bf v; uint4 q[2]; };

#ifndef __has_builtin
#define __has_builtin(x) 0
#endif

#if __has_builtin(__builtin_amdgcn_global_load_async_to_lds_b128)
#define HAVE_ASYNC_LDS 1
#else
#define HAVE_ASYNC_LDS 0
#endif

#if HAVE_ASYNC_LDS
typedef int v4i_t __attribute__((vector_size(16)));
typedef __attribute__((address_space(1))) v4i_t glb_v4i;
typedef __attribute__((address_space(3))) v4i_t lds_v4i;
// async copy 16B global -> LDS, tracked by ASYNCcnt (per-lane, like VMEM)
__device__ __forceinline__ void async_cp16(unsigned short* l, const unsigned short* g) {
    __builtin_amdgcn_global_load_async_to_lds_b128((glb_v4i*)g, (lds_v4i*)l, 0, 0);
}
__device__ __forceinline__ void async_wait() {
#if __has_builtin(__builtin_amdgcn_s_wait_asynccnt)
    __builtin_amdgcn_s_wait_asynccnt(0);
#else
    asm volatile("s_wait_asynccnt 0x0" ::: "memory");
#endif
}
#else
__device__ __forceinline__ void async_cp16(unsigned short* l, const unsigned short* g) {
    *(uint4*)l = *(const uint4*)g;
}
__device__ __forceinline__ void async_wait() {}
#endif

__device__ __forceinline__ unsigned short f2bf_bits(float f) {
    union { float f; unsigned u; } c; c.f = f;
    unsigned u = c.u;
    unsigned r = u + 0x7FFFu + ((u >> 16) & 1u);   // round-to-nearest-even
    return (unsigned short)(r >> 16);
}

// Load one 16x16x32 bf16 A/B fragment from LDS (row-major, ld in elements).
// Per ISA: lane half h holds K elements {8h..8h+7} U {16+8h..16+8h+7}.
__device__ __forceinline__ v16bf load_frag(const unsigned short* base, int row,
                                           int ld, int kOff, int h) {
    FragBF f;
    const unsigned short* p = base + row * ld + kOff + 8 * h;
    f.q[0] = *(const uint4*)(p);
    f.q[1] = *(const uint4*)(p + 16);
    return f.v;
}

// ---------------------------------------------------------------- converts
__global__ void convert_f32_bf16(const float* __restrict__ src,
                                 unsigned short* __restrict__ dst, int n) {
    int i = blockIdx.x * blockDim.x + threadIdx.x;
    if (i < n) dst[i] = f2bf_bits(src[i]);
}

// src [K][Nn] fp32 -> dst [Nn][K] bf16  (B^T so WMMA B-fragments are row reads)
__global__ void transpose_f32_bf16(const float* __restrict__ src,
                                   unsigned short* __restrict__ dst,
                                   int K, int Nn) {
    int i = blockIdx.x * blockDim.x + threadIdx.x;
    if (i < K * Nn) {
        int k = i / Nn, n = i % Nn;
        dst[(size_t)n * K + k] = f2bf_bits(src[i]);
    }
}

// ---------------------------------------------------------------- bf16 GEMM
// C[M][Nn](fp32) = A[M][K](bf16) * Bt[Nn][K]^T(bf16) + bias
// Tile 128x128xBK32; 8 waves; wave (wm,wn) owns 32 rows x 64 cols = 8 WMMAs.
// Async double-buffered LDS staging.
__launch_bounds__(256)
__global__ void gemm_bf16(const unsigned short* __restrict__ A,
                          const unsigned short* __restrict__ Bt,
                          const float* __restrict__ bias,
                          float* __restrict__ Cmat,
                          int M, int Nn, int K) {
    const int LDT = 40;                        // 32 + 8 pad (conflict-free b128)
    __shared__ unsigned short As[2][128 * 40];
    __shared__ unsigned short Bs[2][128 * 40];

    const int tid = threadIdx.x;
    const int w = tid >> 5, lane = tid & 31;
    const int hl = lane >> 4, lm = lane & 15;
    const int wm = w & 3;        // 4 M-groups of 32 rows
    const int wn = w >> 2;       // 2 N-groups of 64 cols
    const int m0 = blockIdx.x * 128;
    const int n0 = blockIdx.y * 128;

    // branch-free staging: every thread copies exactly 2 A + 2 B 16B chunks
    auto stage = [&](int buf, int k0) {
#pragma unroll
        for (int j = 0; j < 2; ++j) {
            int i = tid + j * 256;             // 0..511
            int r = i >> 2, cc = i & 3;
            async_cp16(&As[buf][r * LDT + cc * 8],
                       A + (size_t)(m0 + r) * K + k0 + cc * 8);
        }
#pragma unroll
        for (int j = 0; j < 2; ++j) {
            int i = tid + j * 256;
            int r = i >> 2, cc = i & 3;
            async_cp16(&Bs[buf][r * LDT + cc * 8],
                       Bt + (size_t)(n0 + r) * K + k0 + cc * 8);
        }
    };

    v8f acc[2][4];
#pragma unroll
    for (int i = 0; i < 2; ++i)
#pragma unroll
        for (int t = 0; t < 4; ++t)
#pragma unroll
            for (int r = 0; r < 8; ++r) acc[i][t][r] = 0.0f;

    stage(0, 0);
    async_wait();
    __syncthreads();

    for (int k0 = 0; k0 < K; k0 += 32) {
        const int cur = (k0 >> 5) & 1;
        if (k0 + 32 < K) stage(cur ^ 1, k0 + 32);   // overlap with compute

        v16bf a0 = load_frag(As[cur], wm * 32 + lm,      LDT, 0, hl);
        v16bf a1 = load_frag(As[cur], wm * 32 + 16 + lm, LDT, 0, hl);
#pragma unroll
        for (int t = 0; t < 4; ++t) {
            v16bf b = load_frag(Bs[cur], wn * 64 + t * 16 + lm, LDT, 0, hl);
            acc[0][t] = __builtin_amdgcn_wmma_f32_16x16x32_bf16(
                false, a0, false, b, (short)0, acc[0][t], false, false);
            acc[1][t] = __builtin_amdgcn_wmma_f32_16x16x32_bf16(
                false, a1, false, b, (short)0, acc[1][t], false, false);
        }
        async_wait();
        __syncthreads();
    }

    // epilogue: C layout — reg r, lane half hl: M = r + 8*hl, N = lm
#pragma unroll
    for (int i = 0; i < 2; ++i)
#pragma unroll
        for (int t = 0; t < 4; ++t)
#pragma unroll
            for (int r = 0; r < 8; ++r) {
                int m = m0 + wm * 32 + i * 16 + r + 8 * hl;
                int n = n0 + wn * 64 + t * 16 + lm;
                Cmat[(size_t)m * Nn + n] = acc[i][t][r] + bias[n];
            }
}

// ------------------------------------------------- RMSNorm + RoPE epilogue
// qkv fp32 [N][3*C] -> Qb,Kb bf16 [H][N][D], Vt bf16 [H][D][N]
__global__ void qkv_post(const float* __restrict__ qkv,
                         const int* __restrict__ coords,
                         const float* __restrict__ gq,
                         const float* __restrict__ gk,
                         unsigned short* __restrict__ Qb,
                         unsigned short* __restrict__ Kb,
                         unsigned short* __restrict__ Vt) {
    int idx = blockIdx.x * 256 + threadIdx.x;   // 65536 = N*H
    int h = idx & 15, n = idx >> 4;

    // RoPE phases: 3 axes x FREQ_DIM=10, + 2 pad (cos=1,sin=0)
    float cs[32], sn[32];
    int cc[3] = {coords[n * 3], coords[n * 3 + 1], coords[n * 3 + 2]};
#pragma unroll
    for (int a = 0; a < 3; ++a)
#pragma unroll
        for (int fi = 0; fi < 10; ++fi) {
            float freq = __expf(-(float)fi * 0.1f * 9.210340371976184f);
            float ph = (float)cc[a] * freq;
            cs[a * 10 + fi] = cosf(ph);
            sn[a * 10 + fi] = sinf(ph);
        }
    cs[30] = 1.f; cs[31] = 1.f; sn[30] = 0.f; sn[31] = 0.f;

    const float* qrow = qkv + (size_t)n * C3 + h * DH;
    const float* krow = qrow + CDIM;
    const float* vrow = qrow + 2 * CDIM;
    const float* gqr = gq + h * DH;
    const float* gkr = gk + h * DH;
    unsigned short* qo = Qb + ((size_t)h * NTOK + n) * DH;
    unsigned short* ko = Kb + ((size_t)h * NTOK + n) * DH;

    {
        float ssq = 0.f;
#pragma unroll
        for (int d = 0; d < DH; ++d) { float v = qrow[d]; ssq += v * v; }
        float inv = 8.0f / fmaxf(sqrtf(ssq), 1e-12f);   // sqrt(D)/||t||
#pragma unroll
        for (int p = 0; p < 32; ++p) {
            float t0 = qrow[2 * p]     * inv * gqr[2 * p];
            float t1 = qrow[2 * p + 1] * inv * gqr[2 * p + 1];
            qo[2 * p]     = f2bf_bits(t0 * cs[p] - t1 * sn[p]);
            qo[2 * p + 1] = f2bf_bits(t0 * sn[p] + t1 * cs[p]);
        }
    }
    {
        float ssq = 0.f;
#pragma unroll
        for (int d = 0; d < DH; ++d) { float v = krow[d]; ssq += v * v; }
        float inv = 8.0f / fmaxf(sqrtf(ssq), 1e-12f);
#pragma unroll
        for (int p = 0; p < 32; ++p) {
            float t0 = krow[2 * p]     * inv * gkr[2 * p];
            float t1 = krow[2 * p + 1] * inv * gkr[2 * p + 1];
            ko[2 * p]     = f2bf_bits(t0 * cs[p] - t1 * sn[p]);
            ko[2 * p + 1] = f2bf_bits(t0 * sn[p] + t1 * cs[p]);
        }
    }
#pragma unroll
    for (int d = 0; d < DH; ++d)
        Vt[((size_t)h * DH + d) * NTOK + n] = f2bf_bits(vrow[d]);
}

// ------------------------------------------------------- flash attention
// grid (H, N/64); block 128 threads = 4 waves; wave w owns q-rows [16w,16w+16)
// Async double-buffered K/V tiles of 128 keys.
__launch_bounds__(128)
__global__ void flash_attn(const unsigned short* __restrict__ Qb,
                           const unsigned short* __restrict__ Kb,
                           const unsigned short* __restrict__ Vt,
                           unsigned short* __restrict__ hb) {
    const int LQ = 72;    // 64 + 8 pad
    const int LP = 136;   // 128 + 8 pad
    __shared__ unsigned short Qs[64 * 72];        //  9216 B
    __shared__ unsigned short Ks[2][128 * 72];    // 36864 B
    __shared__ unsigned short Vs[2][64 * 136];    // 34816 B
    __shared__ unsigned short Ps[64 * 136];       // 17408 B (total 98304 B)

    const int h = blockIdx.x;
    const int qb0 = blockIdx.y * 64;
    const int tid = threadIdx.x;
    const int w = tid >> 5, lane = tid & 31;
    const int hl = lane >> 4, lm = lane & 15;

    const unsigned short* Qg = Qb + ((size_t)h * NTOK + qb0) * DH;
    const unsigned short* Kg = Kb + (size_t)h * NTOK * DH;
    const unsigned short* Vg = Vt + (size_t)h * DH * NTOK;

    auto stageKV = [&](int buf, int kb) {
#pragma unroll
        for (int j = 0; j < 8; ++j) {               // K: 128x64 = 1024 chunks
            int i = tid + j * 128;
            int r = i >> 3, cc = i & 7;
            async_cp16(&Ks[buf][r * LQ + cc * 8],
                       Kg + (size_t)(kb + r) * DH + cc * 8);
        }
#pragma unroll
        for (int j = 0; j < 8; ++j) {               // V^T: 64x128 = 1024 chunks
            int i = tid + j * 128;
            int r = i >> 4, cc = i & 15;
            async_cp16(&Vs[buf][r * LP + cc * 8],
                       Vg + (size_t)r * NTOK + kb + cc * 8);
        }
    };

    // stage Q tile: 64x64 = 512 chunks, 4 per thread
#pragma unroll
    for (int j = 0; j < 4; ++j) {
        int i = tid + j * 128;
        int r = i >> 3, cc = i & 7;
        async_cp16(&Qs[r * LQ + cc * 8], Qg + (size_t)r * DH + cc * 8);
    }

    v8f o[4];
    float mrow[8], lrow[8];
#pragma unroll
    for (int t = 0; t < 4; ++t)
#pragma unroll
        for (int r = 0; r < 8; ++r) o[t][r] = 0.0f;
#pragma unroll
    for (int r = 0; r < 8; ++r) { mrow[r] = -1e30f; lrow[r] = 0.0f; }

    stageKV(0, 0);
    async_wait();
    __syncthreads();

    for (int kb = 0; kb < NTOK; kb += 128) {
        const int cur = (kb >> 7) & 1;
        if (kb + 128 < NTOK) stageKV(cur ^ 1, kb + 128);   // overlap

        // S = (Q K^T) / sqrt(D): 8 n-subtiles x 2 K-chunks of 32
        float sv[8][8];
#pragma unroll
        for (int nt = 0; nt < 8; ++nt) {
            v8f s;
#pragma unroll
            for (int r = 0; r < 8; ++r) s[r] = 0.0f;
#pragma unroll
            for (int kc = 0; kc < 2; ++kc) {
                v16bf a = load_frag(Qs, w * 16 + lm, LQ, kc * 32, hl);
                v16bf b = load_frag(Ks[cur], nt * 16 + lm, LQ, kc * 32, hl);
                s = __builtin_amdgcn_wmma_f32_16x16x32_bf16(
                    false, a, false, b, (short)0, s, false, false);
            }
#pragma unroll
            for (int r = 0; r < 8; ++r) sv[nt][r] = s[r] * 0.125f;
        }

        // online softmax (row stats replicated across each 16-lane half)
        float alpha[8];
#pragma unroll
        for (int r = 0; r < 8; ++r) {
            float mx = sv[0][r];
#pragma unroll
            for (int nt = 1; nt < 8; ++nt) mx = fmaxf(mx, sv[nt][r]);
#pragma unroll
            for (int off = 1; off < 16; off <<= 1)
                mx = fmaxf(mx, __shfl_xor(mx, off, 16));
            float mnew = fmaxf(mrow[r], mx);
            alpha[r] = __expf(mrow[r] - mnew);
            mrow[r] = mnew;
            float rs = 0.f;
#pragma unroll
            for (int nt = 0; nt < 8; ++nt) {
                float p = __expf(sv[nt][r] - mnew);
                sv[nt][r] = p;
                rs += p;
            }
#pragma unroll
            for (int off = 1; off < 16; off <<= 1)
                rs += __shfl_xor(rs, off, 16);
            lrow[r] = lrow[r] * alpha[r] + rs;
        }

        // emit P (bf16) into this wave's private LDS rows (same-wave DS in-order)
#pragma unroll
        for (int nt = 0; nt < 8; ++nt)
#pragma unroll
            for (int r = 0; r < 8; ++r)
                Ps[(w * 16 + r + 8 * hl) * LP + nt * 16 + lm] = f2bf_bits(sv[nt][r]);

#pragma unroll
        for (int t = 0; t < 4; ++t)
#pragma unroll
            for (int r = 0; r < 8; ++r) o[t][r] *= alpha[r];

        // O += P V : K = 128 -> 4 chunks of 32
#pragma unroll
        for (int kc = 0; kc < 4; ++kc) {
            v16bf a = load_frag(Ps, w * 16 + lm, LP, kc * 32, hl);
#pragma unroll
            for (int t = 0; t < 4; ++t) {
                v16bf b = load_frag(Vs[cur], t * 16 + lm, LP, kc * 32, hl);
                o[t] = __builtin_amdgcn_wmma_f32_16x16x32_bf16(
                    false, a, false, b, (short)0, o[t], false, false);
            }
        }
        async_wait();
        __syncthreads();
    }

    // write hb[n][h*64 + c] bf16, normalized by row sum
#pragma unroll
    for (int t = 0; t < 4; ++t)
#pragma unroll
        for (int r = 0; r < 8; ++r) {
            int m = qb0 + w * 16 + r + 8 * hl;
            int c = t * 16 + lm;
            hb[(size_t)m * CDIM + h * DH + c] = f2bf_bits(o[t][r] / lrow[r]);
        }
}

// ---------------------------------------------------------------- launcher
extern "C" void kernel_launch(void* const* d_in, const int* in_sizes, int n_in,
                              void* d_out, int out_size, void* d_ws, size_t ws_size,
                              hipStream_t stream) {
    const float* x      = (const float*)d_in[0];
    const int*   coords = (const int*)  d_in[1];
    const float* Wqkv   = (const float*)d_in[2];
    const float* bqkv   = (const float*)d_in[3];
    const float* gq     = (const float*)d_in[4];
    const float* gk     = (const float*)d_in[5];
    const float* Wout   = (const float*)d_in[6];
    const float* bout   = (const float*)d_in[7];
    float* out = (float*)d_out;

    char* ws = (char*)d_ws;
    unsigned short* xb     = (unsigned short*)(ws);                  //  8 MB
    unsigned short* WqkvT  = (unsigned short*)(ws + (8u  << 20));    //  6 MB
    unsigned short* WoutT  = (unsigned short*)(ws + (14u << 20));    //  2 MB
    float*          qkv    = (float*)         (ws + (16u << 20));    // 48 MB
    unsigned short* Qb     = (unsigned short*)(ws + (64u << 20));    //  8 MB
    unsigned short* Kb     = (unsigned short*)(ws + (72u << 20));    //  8 MB
    unsigned short* Vt     = (unsigned short*)(ws + (80u << 20));    //  8 MB
    unsigned short* hb     = (unsigned short*)(ws + (88u << 20));    //  8 MB -> 96 MB total

    convert_f32_bf16<<<(NTOK * CDIM) / 256, 256, 0, stream>>>(x, xb, NTOK * CDIM);
    transpose_f32_bf16<<<(CDIM * C3) / 256, 256, 0, stream>>>(Wqkv, WqkvT, CDIM, C3);
    transpose_f32_bf16<<<(CDIM * CDIM) / 256, 256, 0, stream>>>(Wout, WoutT, CDIM, CDIM);

    gemm_bf16<<<dim3(NTOK / 128, C3 / 128), 256, 0, stream>>>(
        xb, WqkvT, bqkv, qkv, NTOK, C3, CDIM);

    qkv_post<<<(NTOK * HDS) / 256, 256, 0, stream>>>(qkv, coords, gq, gk, Qb, Kb, Vt);

    flash_attn<<<dim3(HDS, NTOK / 64), 128, 0, stream>>>(Qb, Kb, Vt, hb);

    gemm_bf16<<<dim3(NTOK / 128, CDIM / 128), 256, 0, stream>>>(
        hb, WoutT, bout, out, NTOK, CDIM, CDIM);
}
